// GenMid_55800215109840
// MI455X (gfx1250) — matrix operand order repaired and checked by reading, not compiled, not run on hardware
//
#include <hip/hip_runtime.h>
#include <math.h>

// ---------------------------------------------------------------------------
// Shapes
//   z(8,100) @ w_l1(100,8192) -> BN1 -> up2x -> I(8,128,16,16) const LIF input
//   LIF 200 steps -> spikes (binary) -> conv1 as WMMA GEMM M=128,N=2048,K=230400
//   BN2+leaky -> up2x -> conv2 WMMA GEMM M=64,N=8192,K=1152 -> BN3+leaky
//   conv3 (3 out ch) + tanh -> out (8,3,32,32)
// ---------------------------------------------------------------------------

typedef __attribute__((ext_vector_type(16))) __bf16 v16bf;
typedef __attribute__((ext_vector_type(8)))  float  v8f;
typedef int v4i __attribute__((vector_size(16)));

union FragB16 { float4 f4[2]; v16bf v; };

__device__ __forceinline__ unsigned short f32_bf16(float f) {
  unsigned int u = __float_as_uint(f);
  u += 0x7FFFu + ((u >> 16) & 1u);          // round-to-nearest-even
  return (unsigned short)(u >> 16);
}

// ---- CDNA5 async global->LDS copy (ASYNCcnt) ------------------------------
#if __has_builtin(__builtin_amdgcn_global_load_async_to_lds_b128)
#define HAVE_ASYNC_LDS 1
__device__ __forceinline__ void async_cp16(const void* g, void* l) {
  __builtin_amdgcn_global_load_async_to_lds_b128((v4i*)g, (v4i*)l, 0, 0);
}
#else
#define HAVE_ASYNC_LDS 0
#endif

#if __has_builtin(__builtin_amdgcn_s_wait_asynccnt)
#define WAIT_ASYNC(n) __builtin_amdgcn_s_wait_asynccnt(n)
#else
#define WAIT_ASYNC(n) asm volatile("s_wait_asynccnt %0" ::"n"(n) : "memory")
#endif

// ----------------------------- workspace layout ----------------------------
// spkT  : bf16 [b0(8)][hh(18)][ww(18)][t*128+c (25600)]  (halo = zero)
// c1    : f32  [o(128)][pix(2048)]   (atomic accumulate -> must be zeroed)
// x2T   : bf16 [b(8)][hh(34)][ww(34)][o(128)]            (halo = zero)
static constexpr long long OFF_SPK = 0;                       // 132,710,400 B
static constexpr long long SPK_B   = (long long)8*18*18*25600*2;
static constexpr long long OFF_C1  = OFF_SPK + SPK_B;         // 1,048,576 B
static constexpr long long OFF_X2T = OFF_C1 + (long long)128*2048*4;
static constexpr long long X2T_B   = (long long)8*34*34*128*2;
static constexpr long long ZERO_END= OFF_X2T + X2T_B;         // 136,126,464 B
static constexpr long long OFF_LIN = ZERO_END;                // 8*8192 f32
static constexpr long long OFF_I   = OFF_LIN + (long long)8*8192*4;
static constexpr long long OFF_W1  = OFF_I   + (long long)8*128*256*4; // bf16 [9][128][25600]
static constexpr long long OFF_W2  = OFF_W1  + (long long)9*128*25600*2; // bf16 [9][64][128]
static constexpr long long OFF_C2  = OFF_W2  + (long long)9*64*128*2;  // f32 [64][8192]
static constexpr long long OFF_Y3  = OFF_C2  + (long long)64*8192*4;   // f32 [8][64][32][32]
static constexpr long long WS_NEED = OFF_Y3  + (long long)8*64*1024*4; // ~191.5 MB

// ------------------------------ small kernels ------------------------------
__global__ void k_zero(uint4* p, long long n) {
  long long i = (long long)blockIdx.x * blockDim.x + threadIdx.x;
  long long stride = (long long)gridDim.x * blockDim.x;
  uint4 zv; zv.x = zv.y = zv.z = zv.w = 0u;
  for (; i < n; i += stride) p[i] = zv;
}

// w_c1 (o,25600,3,3) f32 -> Wt1 [kk][o][tc] bf16
__global__ void k_prep_w1(const float* __restrict__ w, unsigned short* __restrict__ Wt) {
  int idx = blockIdx.x * blockDim.x + threadIdx.x;           // 9*128*25600
  int tc = idx % 25600;
  int r  = idx / 25600;
  int o  = r % 128;
  int kk = r / 128;
  Wt[idx] = f32_bf16(w[(size_t)(o * 25600 + tc) * 9 + kk]);
}

// w_c2 (64,128,3,3) f32 -> Wt2 [kk][o][c] bf16
__global__ void k_prep_w2(const float* __restrict__ w, unsigned short* __restrict__ Wt) {
  int idx = blockIdx.x * blockDim.x + threadIdx.x;           // 9*64*128
  int c  = idx & 127;
  int r  = idx >> 7;
  int o  = r & 63;
  int kk = r >> 6;
  Wt[idx] = f32_bf16(w[(size_t)(o * 128 + c) * 9 + kk]);
}

// lin[b][j] = z(b,:) . w_l1(:,j) + b_l1[j]
__global__ void k_linear(const float* __restrict__ z, const float* __restrict__ w,
                         const float* __restrict__ bias, float* __restrict__ lin) {
  int idx = blockIdx.x * blockDim.x + threadIdx.x;           // 8*8192
  int j = idx & 8191, b = idx >> 13;
  float s = bias[j];
  for (int k = 0; k < 100; k++) s += z[b * 100 + k] * w[(size_t)k * 8192 + j];
  lin[idx] = s;
}

// BN1 (training stats over N,H,W) + nearest up2x -> I (8,128,16,16)
__global__ void k_bn1(const float* __restrict__ lin, const float* __restrict__ g1,
                      const float* __restrict__ be1, float* __restrict__ I) {
  int c = blockIdx.x, tid = threadIdx.x;
  __shared__ float s1[256], s2[256];
  float sum = 0.f, sq = 0.f;
  for (int s = tid; s < 512; s += 256) {
    int b = s >> 6, i = s & 63;
    float v = lin[b * 8192 + c * 64 + i];
    sum += v; sq += v * v;
  }
  s1[tid] = sum; s2[tid] = sq; __syncthreads();
  for (int st = 128; st > 0; st >>= 1) {
    if (tid < st) { s1[tid] += s1[tid + st]; s2[tid] += s2[tid + st]; }
    __syncthreads();
  }
  float mean = s1[0] / 512.f;
  float var  = s2[0] / 512.f - mean * mean;
  float scale = g1[c] * rsqrtf(var + 1e-5f);
  float shift = be1[c] - mean * scale;
  for (int s = tid; s < 512; s += 256) {
    int b = s >> 6, i = s & 63;
    int h0 = i >> 3, w0 = i & 7;
    float v = lin[b * 8192 + c * 64 + i] * scale + shift;
    float* base = I + ((size_t)(b * 128 + c) * 16) * 16;
    base[(2*h0    ) * 16 + 2*w0    ] = v;
    base[(2*h0    ) * 16 + 2*w0 + 1] = v;
    base[(2*h0 + 1) * 16 + 2*w0    ] = v;
    base[(2*h0 + 1) * 16 + 2*w0 + 1] = v;
  }
}

// LIF 200 steps, reset-by-subtraction; store bf16 spikes pixel-major w/ halo.
__global__ void k_lif(const float* __restrict__ I, unsigned short* __restrict__ spk) {
  int c = threadIdx.x;
  int blk = blockIdx.x;                  // b*256 + h*16 + w
  int b = blk >> 8, h = (blk >> 4) & 15, w = blk & 15;
  float Iv = I[((size_t)(b * 128 + c) * 16 + h) * 16 + w];
  unsigned short* dst = spk + ((size_t)((b * 18 + h + 1) * 18) + (w + 1)) * 25600 + c;
  float mem = 0.f;
  for (int t = 0; t < 200; t++) {
    float reset = (mem > 1.0f) ? 1.0f : 0.0f;
    mem = 0.95f * mem + Iv - reset;
    dst[(size_t)t * 128] = (mem > 1.0f) ? (unsigned short)0x3F80 : (unsigned short)0;
  }
}

// --------------------------- conv1 WMMA GEMM -------------------------------
// M=128 (o), N=2048 (b',h,w), K=9 x 25600.  WG = 8 waves, tile 128M x 128N.
// grid = (16 N-groups, 20 K-splits).  WG's 128 pixels sit inside one image
// (b' = blk>>1), so the torch-reshape quirk (t8b = b'*200 + t0/128) is
// wave-uniform: per step only scalar bases change.
//
// Async variant: triple-buffered LDS (48KB), GLOBAL_LOAD_ASYNC_TO_LDS_B128
// staging tracked by ASYNCcnt; one barrier per K-step:
//   wait asynccnt<=4 ; barrier ; stage(s+2) ; consume(s)
// (buffer (s+2)%3 was last read at step s-1, so the barrier makes the
//  write-after-read safe; async loads complete in order per wave.)
__global__ void __launch_bounds__(256)
k_conv1(const unsigned short* __restrict__ Wt1,
        const unsigned short* __restrict__ spk, float* __restrict__ c1) {
#if HAVE_ASYNC_LDS
  __shared__ alignas(16) unsigned short ldsA[3][4096];   // [m(128)][k(32)]
  __shared__ alignas(16) unsigned short ldsB[3][4096];   // [n(128)][k(32)]
#else
  __shared__ alignas(16) unsigned short ldsA[2][4096];
  __shared__ alignas(16) unsigned short ldsB[2][4096];
#endif

  const int tid  = threadIdx.x;
  const int lane = tid & 31;
  const int wv   = tid >> 5;
  const int mrow = wv & 3;                 // 4 M-groups of 32
  const int ncol = wv >> 2;                // 2 N-groups of 64
  const int lm   = lane & 15;
  const int hi   = (lane & 16) ? 1 : 0;

  const int blk   = blockIdx.x;            // 16 N-groups of 128 pixels
  const int tcs   = blockIdx.y * 1280;     // 20 K-splits (10 blocks of 128)
  const int bimg  = blk >> 1;              // image batch b' (WG-uniform)
  const int b200  = bimg * 200;
  const int hbase = (blk & 1) * 8;

  // thread-constant offsets: 2 A units + 2 B units (float4 = 8 bf16) / thread
  size_t aoffg[2], boffg[2];
  int    aoffl[2], boffl[2];
#pragma unroll
  for (int j = 0; j < 2; j++) {
    int u  = tid + j * 256;
    int rw = u >> 2, kc = u & 3;           // row (m or n), k-chunk of 8
    aoffg[j] = (size_t)rw * 25600 + kc * 8;
    aoffl[j] = rw * 32 + kc * 8;
    int hh = hbase + (rw >> 4);
    int wn = rw & 15;
    boffg[j] = (size_t)(hh * 18 + wn) * 25600 + kc * 8;
    boffl[j] = rw * 32 + kc * 8;
  }

  v8f acc[2][4];
  v8f z8 = {0.f,0.f,0.f,0.f,0.f,0.f,0.f,0.f};
#pragma unroll
  for (int i = 0; i < 2; i++)
#pragma unroll
    for (int j = 0; j < 4; j++) acc[i][j] = z8;

  auto bases = [&](int kk, int kh, int kw, int t0,
                   const unsigned short*& pa, const unsigned short*& pb) {
    pa = Wt1 + (size_t)kk * 3276800 + t0;            // 128*25600
    int t8bl = t0 >> 7, c0 = t0 & 127;
    int t8b  = b200 + t8bl;
    pb = spk + (size_t)(((t8b & 7) * 18 + kh) * 18 + kw) * 25600
             + (t8b >> 3) * 128 + c0;
  };

  const int NSTEPS = 360;                  // 9 (kh,kw) * 40 K-steps of 32

#if HAVE_ASYNC_LDS
  // ---- async triple-buffered pipeline ----
  auto stage = [&](int kk, int kh, int kw, int t0, int buf) {
    const unsigned short *pa, *pb;
    bases(kk, kh, kw, t0, pa, pb);
    async_cp16(pa + aoffg[0], &ldsA[buf][aoffl[0]]);
    async_cp16(pa + aoffg[1], &ldsA[buf][aoffl[1]]);
    async_cp16(pb + boffg[0], &ldsB[buf][boffl[0]]);
    async_cp16(pb + boffg[1], &ldsB[buf][boffl[1]]);
  };

  stage(0, 0, 0, tcs,      0);             // step 0
  stage(0, 0, 0, tcs + 32, 1);             // step 1
  int pkk = 0, pkh = 0, pkw = 0, pts = 2;  // prefetch position = step 2
  int bufw = 2, bufr = 0;

#pragma unroll 3
  for (int s = 0; s < NSTEPS; s++) {
    if (s < NSTEPS - 1) WAIT_ASYNC(4); else WAIT_ASYNC(0);
    __syncthreads();
    if (s < NSTEPS - 2) {
      stage(pkk, pkh, pkw, tcs + pts * 32, bufw);
      if (++pts == 40) { pts = 0; pkk++; if (++pkw == 3) { pkw = 0; pkh++; } }
      if (++bufw == 3) bufw = 0;
    }
    const unsigned short* lA = ldsA[bufr];
    const unsigned short* lB = ldsB[bufr];
    if (++bufr == 3) bufr = 0;
    FragB16 a[2], b[4];
#pragma unroll
    for (int mi = 0; mi < 2; mi++) {
      const unsigned short* p = lA + (mrow * 32 + mi * 16 + lm) * 32 + hi * 8;
      a[mi].f4[0] = *(const float4*)(p);
      a[mi].f4[1] = *(const float4*)(p + 16);
    }
#pragma unroll
    for (int ni = 0; ni < 4; ni++) {
      const unsigned short* p = lB + (ncol * 64 + ni * 16 + lm) * 32 + hi * 8;
      b[ni].f4[0] = *(const float4*)(p);
      b[ni].f4[1] = *(const float4*)(p + 16);
    }
#pragma unroll
    for (int mi = 0; mi < 2; mi++)
#pragma unroll
      for (int ni = 0; ni < 4; ni++)
        acc[mi][ni] = __builtin_amdgcn_wmma_f32_16x16x32_bf16(
            false, a[mi].v, false, b[ni].v, (short)0, acc[mi][ni],
            false, false);
  }
#else
  // ---- fallback: reg-staged double-buffered pipeline ----
  float4 r0, r1, r2, r3;
  {
    const unsigned short *pa, *pb;
    bases(0, 0, 0, tcs, pa, pb);
    r0 = *(const float4*)(pa + aoffg[0]);
    r1 = *(const float4*)(pa + aoffg[1]);
    r2 = *(const float4*)(pb + boffg[0]);
    r3 = *(const float4*)(pb + boffg[1]);
  }
  *(float4*)(&ldsA[0][aoffl[0]]) = r0;
  *(float4*)(&ldsA[0][aoffl[1]]) = r1;
  *(float4*)(&ldsB[0][boffl[0]]) = r2;
  *(float4*)(&ldsB[0][boffl[1]]) = r3;
  __syncthreads();

  int kk = 0, kh = 0, kw = 0, tstep = 0;
#pragma unroll 2
  for (int s = 0; s < NSTEPS; s++) {
    const int cur = s & 1, nxt = cur ^ 1;
    int kk2 = kk, kh2 = kh, kw2 = kw, ts2 = tstep + 1;
    if (ts2 == 40) { ts2 = 0; kk2++; kw2++; if (kw2 == 3) { kw2 = 0; kh2++; } }
    const bool more = (s + 1) < NSTEPS;
    if (more) {
      const unsigned short *pa, *pb;
      bases(kk2, kh2, kw2, tcs + ts2 * 32, pa, pb);
      r0 = *(const float4*)(pa + aoffg[0]);
      r1 = *(const float4*)(pa + aoffg[1]);
      r2 = *(const float4*)(pb + boffg[0]);
      r3 = *(const float4*)(pb + boffg[1]);
    }
    const unsigned short* lA = ldsA[cur];
    const unsigned short* lB = ldsB[cur];
    FragB16 a[2], b[4];
#pragma unroll
    for (int mi = 0; mi < 2; mi++) {
      const unsigned short* p = lA + (mrow * 32 + mi * 16 + lm) * 32 + hi * 8;
      a[mi].f4[0] = *(const float4*)(p);
      a[mi].f4[1] = *(const float4*)(p + 16);
    }
#pragma unroll
    for (int ni = 0; ni < 4; ni++) {
      const unsigned short* p = lB + (ncol * 64 + ni * 16 + lm) * 32 + hi * 8;
      b[ni].f4[0] = *(const float4*)(p);
      b[ni].f4[1] = *(const float4*)(p + 16);
    }
#pragma unroll
    for (int mi = 0; mi < 2; mi++)
#pragma unroll
      for (int ni = 0; ni < 4; ni++)
        acc[mi][ni] = __builtin_amdgcn_wmma_f32_16x16x32_bf16(
            false, a[mi].v, false, b[ni].v, (short)0, acc[mi][ni],
            false, false);
    if (more) {
      *(float4*)(&ldsA[nxt][aoffl[0]]) = r0;
      *(float4*)(&ldsA[nxt][aoffl[1]]) = r1;
      *(float4*)(&ldsB[nxt][boffl[0]]) = r2;
      *(float4*)(&ldsB[nxt][boffl[1]]) = r3;
    }
    __syncthreads();
    kk = kk2; kh = kh2; kw = kw2; tstep = ts2;
  }
#endif

  // epilogue: D vgpr r -> (M = r + 8*hi, N = lane%16); K-split combine
#pragma unroll
  for (int mi = 0; mi < 2; mi++)
#pragma unroll
    for (int ni = 0; ni < 4; ni++) {
      int p = blk * 128 + ncol * 64 + ni * 16 + lm;
      int ob = mrow * 32 + mi * 16 + (hi ? 8 : 0);
#pragma unroll
      for (int r = 0; r < 8; r++)
        atomicAdd(&c1[(size_t)(ob + r) * 2048 + p], acc[mi][ni][r]);
    }
}

// BN2(eps=0.8)+leaky + nearest up2x -> x2T bf16 [b][34][34][o] (halo zeroed)
__global__ void k_bn2(const float* __restrict__ c1, const float* __restrict__ bc1,
                      const float* __restrict__ g2, const float* __restrict__ be2,
                      unsigned short* __restrict__ x2T) {
  int o = blockIdx.x, tid = threadIdx.x;
  __shared__ float s1[256], s2[256];
  float bias = bc1[o];
  float sum = 0.f, sq = 0.f;
  for (int p = tid; p < 2048; p += 256) {
    float x = c1[(size_t)o * 2048 + p] + bias;
    sum += x; sq += x * x;
  }
  s1[tid] = sum; s2[tid] = sq; __syncthreads();
  for (int st = 128; st > 0; st >>= 1) {
    if (tid < st) { s1[tid] += s1[tid + st]; s2[tid] += s2[tid + st]; }
    __syncthreads();
  }
  float mean = s1[0] / 2048.f;
  float var  = s2[0] / 2048.f - mean * mean;
  float scale = g2[o] * rsqrtf(var + 0.8f);
  float shift = be2[o] - mean * scale;
  for (int p = tid; p < 2048; p += 256) {
    float v = (c1[(size_t)o * 2048 + p] + bias) * scale + shift;
    v = (v >= 0.f) ? v : 0.2f * v;
    unsigned short hv = f32_bf16(v);
    int b = p >> 8, h = (p >> 4) & 15, w = p & 15;
#pragma unroll
    for (int dh = 0; dh < 2; dh++)
#pragma unroll
      for (int dw = 0; dw < 2; dw++)
        x2T[((size_t)(b * 34 + 1 + 2 * h + dh) * 34 + (1 + 2 * w + dw)) * 128 + o] = hv;
  }
}

// --------------------------- conv2 WMMA GEMM -------------------------------
// M=64 (o), N=8192 (b,h,w on 32x32), K=9x128. WG tile 64M x 128N, grid 64 WGs.
__global__ void __launch_bounds__(256)
k_conv2(const unsigned short* __restrict__ Wt2,
        const unsigned short* __restrict__ x2T, float* __restrict__ c2) {
  const int lane = threadIdx.x & 31;
  const int wv   = threadIdx.x >> 5;
  const int mrow = wv & 1;                 // 2 M-groups of 32
  const int ncol = wv >> 1;                // 4 N-groups of 32
  const int pbase = blockIdx.x * 128 + ncol * 32;
  const int lm = lane & 15;
  const int hi = (lane & 16) ? 1 : 0;

  v8f acc[2][2];
  v8f z8 = {0.f,0.f,0.f,0.f,0.f,0.f,0.f,0.f};
#pragma unroll
  for (int i = 0; i < 2; i++)
#pragma unroll
    for (int j = 0; j < 2; j++) acc[i][j] = z8;

  int bimg[2], h0[2], w0[2];
#pragma unroll
  for (int ni = 0; ni < 2; ni++) {
    int p = pbase + ni * 16;
    bimg[ni] = p >> 10; h0[ni] = (p >> 5) & 31; w0[ni] = p & 31;
  }

  for (int kk = 0; kk < 9; kk++) {
    const int kh = kk / 3, kw = kk % 3;
    const unsigned short* Wk = Wt2 + (size_t)kk * 64 * 128;
    for (int c0 = 0; c0 < 128; c0 += 32) {
      FragB16 a[2];
#pragma unroll
      for (int mi = 0; mi < 2; mi++) {
        const unsigned short* pA =
            Wk + (size_t)(mrow * 32 + mi * 16 + lm) * 128 + c0 + (hi ? 8 : 0);
        a[mi].f4[0] = *(const float4*)(pA);
        a[mi].f4[1] = *(const float4*)(pA + 16);
      }
      FragB16 bfr[2];
#pragma unroll
      for (int ni = 0; ni < 2; ni++) {
        const unsigned short* pB = x2T
            + ((size_t)(bimg[ni] * 34 + h0[ni] + kh) * 34 + (w0[ni] + lm + kw)) * 128
            + c0 + (hi ? 8 : 0);
        bfr[ni].f4[0] = *(const float4*)(pB);
        bfr[ni].f4[1] = *(const float4*)(pB + 16);
      }
#pragma unroll
      for (int mi = 0; mi < 2; mi++)
#pragma unroll
        for (int ni = 0; ni < 2; ni++)
          acc[mi][ni] = __builtin_amdgcn_wmma_f32_16x16x32_bf16(
              false, a[mi].v, false, bfr[ni].v, (short)0, acc[mi][ni],
              false, false);
    }
  }
#pragma unroll
  for (int mi = 0; mi < 2; mi++)
#pragma unroll
    for (int ni = 0; ni < 2; ni++) {
      int p = pbase + ni * 16 + lm;
      int ob = mrow * 32 + mi * 16 + (hi ? 8 : 0);
#pragma unroll
      for (int r = 0; r < 8; r++)
        c2[(size_t)(ob + r) * 8192 + p] = acc[mi][ni][r];
    }
}

// BN3(eps=0.8)+leaky -> y3 f32 (8,64,32,32)
__global__ void k_bn3(const float* __restrict__ c2, const float* __restrict__ bc2,
                      const float* __restrict__ g3, const float* __restrict__ be3,
                      float* __restrict__ y3) {
  int o = blockIdx.x, tid = threadIdx.x;
  __shared__ float s1[256], s2[256];
  float bias = bc2[o];
  float sum = 0.f, sq = 0.f;
  for (int p = tid; p < 8192; p += 256) {
    float x = c2[(size_t)o * 8192 + p] + bias;
    sum += x; sq += x * x;
  }
  s1[tid] = sum; s2[tid] = sq; __syncthreads();
  for (int st = 128; st > 0; st >>= 1) {
    if (tid < st) { s1[tid] += s1[tid + st]; s2[tid] += s2[tid + st]; }
    __syncthreads();
  }
  float mean = s1[0] / 8192.f;
  float var  = s2[0] / 8192.f - mean * mean;
  float scale = g3[o] * rsqrtf(var + 0.8f);
  float shift = be3[o] - mean * scale;
  for (int p = tid; p < 8192; p += 256) {
    float v = (c2[(size_t)o * 8192 + p] + bias) * scale + shift;
    v = (v >= 0.f) ? v : 0.2f * v;
    int b = p >> 10, h = (p >> 5) & 31, w = p & 31;
    y3[((size_t)(b * 64 + o) * 32 + h) * 32 + w] = v;
  }
}

// conv3 (64->3) + tanh, direct f32 (28 MFLOP: negligible)
__global__ void k_conv3(const float* __restrict__ y3, const float* __restrict__ w3,
                        const float* __restrict__ b3, float* __restrict__ out) {
  int idx = blockIdx.x * blockDim.x + threadIdx.x;           // 8*3*1024
  if (idx >= 8 * 3 * 1024) return;
  int w = idx & 31, h = (idx >> 5) & 31;
  int co = (idx >> 10) % 3, b = idx / 3072;
  float s = b3[co];
  for (int ci = 0; ci < 64; ci++) {
    const float* src = y3 + ((size_t)(b * 64 + ci) * 32) * 32;
    const float* wk  = w3 + (size_t)(co * 64 + ci) * 9;
#pragma unroll
    for (int kh = 0; kh < 3; kh++) {
      int y = h + kh - 1;
      if (y < 0 || y > 31) continue;
#pragma unroll
      for (int kw = 0; kw < 3; kw++) {
        int x = w + kw - 1;
        if (x < 0 || x > 31) continue;
        s += src[y * 32 + x] * wk[kh * 3 + kw];
      }
    }
  }
  out[idx] = tanhf(s);
}

// ------------------------------- launcher ----------------------------------
extern "C" void kernel_launch(void* const* d_in, const int* in_sizes, int n_in,
                              void* d_out, int out_size, void* d_ws, size_t ws_size,
                              hipStream_t stream) {
  (void)in_sizes; (void)n_in; (void)out_size;
  if (ws_size < (size_t)WS_NEED) return;

  const float* z    = (const float*)d_in[0];
  const float* w_l1 = (const float*)d_in[1];
  const float* b_l1 = (const float*)d_in[2];
  const float* g1   = (const float*)d_in[3];
  const float* be1  = (const float*)d_in[4];
  const float* w_c1 = (const float*)d_in[5];
  const float* b_c1 = (const float*)d_in[6];
  const float* g2   = (const float*)d_in[7];
  const float* be2  = (const float*)d_in[8];
  const float* w_c2 = (const float*)d_in[9];
  const float* b_c2 = (const float*)d_in[10];
  const float* g3   = (const float*)d_in[11];
  const float* be3  = (const float*)d_in[12];
  const float* w_c3 = (const float*)d_in[13];
  const float* b_c3 = (const float*)d_in[14];

  char* ws = (char*)d_ws;
  unsigned short* spk = (unsigned short*)(ws + OFF_SPK);
  float*          c1  = (float*)(ws + OFF_C1);
  unsigned short* x2T = (unsigned short*)(ws + OFF_X2T);
  float*          lin = (float*)(ws + OFF_LIN);
  float*          I   = (float*)(ws + OFF_I);
  unsigned short* Wt1 = (unsigned short*)(ws + OFF_W1);
  unsigned short* Wt2 = (unsigned short*)(ws + OFF_W2);
  float*          c2  = (float*)(ws + OFF_C2);
  float*          y3  = (float*)(ws + OFF_Y3);

  k_zero   <<<4096, 256, 0, stream>>>((uint4*)ws, ZERO_END / 16);
  k_prep_w1<<<(9*128*25600)/256, 256, 0, stream>>>(w_c1, Wt1);
  k_prep_w2<<<(9*64*128)/256, 256, 0, stream>>>(w_c2, Wt2);
  k_linear <<<256, 256, 0, stream>>>(z, w_l1, b_l1, lin);
  k_bn1    <<<128, 256, 0, stream>>>(lin, g1, be1, I);
  k_lif    <<<2048, 128, 0, stream>>>(I, spk);
  k_conv1  <<<dim3(16, 20), 256, 0, stream>>>(Wt1, spk, c1);
  k_bn2    <<<128, 256, 0, stream>>>(c1, b_c1, g2, be2, x2T);
  k_conv2  <<<64, 256, 0, stream>>>(Wt2, x2T, c2);
  k_bn3    <<<64, 256, 0, stream>>>(c2, b_c2, g3, be3, y3);
  k_conv3  <<<96, 256, 0, stream>>>(y3, w_c3, b_c3, (float*)d_out);
}
